// MultiHeadAttention_SingleAxis_33466385170880
// MI455X (gfx1250) — matrix operand order repaired and checked by reading, not compiled, not run on hardware
//
#include <hip/hip_runtime.h>
#include <hip/hip_bf16.h>
#include <math.h>

// Problem constants (match reference)
#define BB 4
#define HH 128
#define WW 128
#define CC 256
#define DM 512
#define NHH 8
#define DHH 64

typedef __bf16 bf16_t;
typedef __attribute__((ext_vector_type(16))) __bf16 v16bf;
typedef __attribute__((ext_vector_type(8)))  __bf16 v8bf;
typedef __attribute__((ext_vector_type(8)))  float  v8f;

__device__ __forceinline__ bf16_t f2bf(float f) {
    unsigned int u = __float_as_uint(f);
    u += 0x7FFFu + ((u >> 16) & 1u);           // round-to-nearest-even
    unsigned short s = (unsigned short)(u >> 16);
    return __builtin_bit_cast(bf16_t, s);
}

// A fragment (16x32 bf16, M x K): lane l holds row (l&15); K chunks of 8 at
// (l>>4)*8 and (l>>4)*8+16  (ISA 7.12.2, 16-bit A-matrix table).
__device__ __forceinline__ v16bf ldsA(const bf16_t* A, int lda, int row, int kbase, int lane) {
    const bf16_t* p = A + row * lda + kbase + ((lane >> 4) << 3);
    v8bf lo = *(const v8bf*)p;
    v8bf hi = *(const v8bf*)(p + 16);
    v16bf r;
#pragma unroll
    for (int i = 0; i < 8; ++i) { r[i] = lo[i]; r[i + 8] = hi[i]; }
    return r;
}

// B fragment (32x16 bf16, K x N), column-major storage Bcm[n*ldb + k]:
// lane l holds column (l&15); K chunks of 8 at (l>>4)*8 and (l>>4)*8+16.
__device__ __forceinline__ v16bf ldsB(const bf16_t* Bcm, int ldb, int col, int kbase, int lane) {
    const bf16_t* p = Bcm + col * ldb + kbase + ((lane >> 4) << 3);
    v8bf lo = *(const v8bf*)p;
    v8bf hi = *(const v8bf*)(p + 16);
    v16bf r;
#pragma unroll
    for (int i = 0; i < 8; ++i) { r[i] = lo[i]; r[i + 8] = hi[i]; }
    return r;
}

__device__ __forceinline__ v8f wmma_bf16(v16bf a, v16bf b, v8f c) {
    return __builtin_amdgcn_wmma_f32_16x16x32_bf16(false, a, false, b, (short)0, c, false, false);
}

// Async copy of a contiguous global region into LDS, 16B per lane per issue,
// 4 KB per block-wide issue. Tracked by ASYNCcnt (no VGPR round-trip).
__device__ __forceinline__ void async_copy_16B(const void* gbase, void* lbase,
                                               int nbytes, int tid) {
    unsigned lo = (unsigned)(uintptr_t)lbase;   // low 32 bits of generic = LDS byte offset
    int iters = nbytes >> 12;                   // / (256 threads * 16B)
    for (int t = 0; t < iters; ++t) {
        unsigned goff = (unsigned)(tid * 16 + t * 4096);
        unsigned loff = lo + goff;
        asm volatile("global_load_async_to_lds_b128 %0, %1, %2"
                     :: "v"(loff), "v"(goff), "s"(gbase)
                     : "memory");
    }
}

__device__ __forceinline__ void wait_async0() {
    asm volatile("s_wait_asynccnt 0" ::: "memory");
}

__device__ __forceinline__ void init_bias(v8f acc[4], const float* bias, int base, int nl) {
#pragma unroll
    for (int nt = 0; nt < 4; ++nt) {
        float v = bias[base + nt * 16 + nl];
#pragma unroll
        for (int i = 0; i < 8; ++i) acc[nt][i] = v;
    }
}

// [128x256] @ [256x64] with A row-major (lda=256) and B col-major (ldb=256).
__device__ __forceinline__ void gemm_qkv(const bf16_t* A, const bf16_t* Bc, v8f acc[4],
                                         int wv, int nl, int lane) {
    for (int kt = 0; kt < 8; ++kt) {
        v16bf a = ldsA(A, 256, wv * 16 + nl, kt * 32, lane);
#pragma unroll
        for (int nt = 0; nt < 4; ++nt) {
            v16bf b = ldsB(Bc, 256, nt * 16 + nl, kt * 32, lane);
            acc[nt] = wmma_bf16(a, b, acc[nt]);
        }
    }
}

// ---------------------------------------------------------------------------
// Kernel 1: one-shot fp32 -> bf16 weight convert + transpose into workspace.
// WqT/WkT/WvT: [512][256] col-major-for-B  (WxT[j*256+c] = Wx[c*512+j])
// WoT:         [256][512]                  (WoT[j*512+k] = Wo[k*256+j])
// Per-head / per-nc tiles become contiguous 32KB / 64KB blocks.
// ---------------------------------------------------------------------------
__global__ __launch_bounds__(256)
void convert_weights(const float* __restrict__ Wq, const float* __restrict__ Wk,
                     const float* __restrict__ Wv, const float* __restrict__ Wo,
                     bf16_t* __restrict__ WqT, bf16_t* __restrict__ WkT,
                     bf16_t* __restrict__ WvT, bf16_t* __restrict__ WoT) {
    int idx = blockIdx.x * 256 + threadIdx.x;    // 4 sections of 131072
    int sec = idx >> 17;
    int r   = idx & 131071;
    if (sec < 3) {
        const float* src = (sec == 0) ? Wq : (sec == 1) ? Wk : Wv;
        bf16_t*      dst = (sec == 0) ? WqT : (sec == 1) ? WkT : WvT;
        int j = r & 511, c = r >> 9;             // r = c*512 + j : read coalesced
        dst[j * 256 + c] = f2bf(src[r]);
    } else {
        int j = r & 255, k = r >> 8;             // r = k*256 + j : read coalesced
        WoT[j * 512 + k] = f2bf(Wo[r]);
    }
}

// ---------------------------------------------------------------------------
// Kernel 2: fused axial MHA. One workgroup = one (b, w) column of 128 H rows.
// 256 threads = 8 waves; wave wv owns output M-tile rows [wv*16, wv*16+16).
// Weight tiles stream in via double-buffered GLOBAL_LOAD_ASYNC_TO_LDS_B128.
// ---------------------------------------------------------------------------
__global__ __launch_bounds__(256, 1)
void axial_mha_fused(const float* __restrict__ x,
                     const bf16_t* __restrict__ WqT, const bf16_t* __restrict__ WkT,
                     const bf16_t* __restrict__ WvT, const bf16_t* __restrict__ WoT,
                     const float* __restrict__ bq, const float* __restrict__ bk,
                     const float* __restrict__ bv, const float* __restrict__ bo,
                     float* __restrict__ out) {
    extern __shared__ char smem_raw[];
    bf16_t* smem = (bf16_t*)smem_raw;

    // LDS carve-up (bf16 elements): 304 KB total
    bf16_t* Xl   = smem;                 // [128][256]  64 KB  (aliased by Wo tile 0/2)
    bf16_t* ctx  = Xl  + 128 * 256;      // [128][512] 128 KB
    bf16_t* Wt0  = ctx + 128 * 512;      // [64][256]   32 KB  (weight tile buf 0)
    bf16_t* Wt1  = Wt0 + 64 * 256;       // [64][256]   32 KB  (weight tile buf 1)
    bf16_t* Qh   = Wt1 + 64 * 256;       // [128][64]   16 KB (row-major)
    bf16_t* Kh   = Qh  + 128 * 64;       // [128][64]   16 KB (row-major == B col-major)
    bf16_t* Vt   = Kh  + 128 * 64;       // [64][128]   16 KB (V transposed: Vt[d][h])
    bf16_t* P    = Qh;                   // [8 waves][16][128] 32 KB, aliases Qh+Kh
    bf16_t* wbuf[2] = {Wt0, Wt1};
    bf16_t* wobuf[2] = {Xl, Wt0};        // 64 KB Wo tiles (Wt0+Wt1 contiguous)

    const int tid  = threadIdx.x;
    const int lane = tid & 31;
    const int wv   = tid >> 5;
    const int nl   = lane & 15;
    const int hl   = lane >> 4;

    const int bx = blockIdx.x / WW;
    const int wx = blockIdx.x % WW;

    // Prefetch head 0's Wq tile while we load & convert x.
    int cur = 0;
    async_copy_16B(WqT, wbuf[cur], 64 * 256 * 2, tid);

    // ---- Stage 1: load x column -> LDS (bf16). Coalesced: one h-row per iter.
    for (int idx = tid; idx < 128 * 256; idx += 256) {
        int h = idx >> 8, c = idx & 255;
        Xl[h * 256 + c] = f2bf(x[(((size_t)bx * HH + h) * WW + wx) * CC + c]);
    }

    // ---- Stage 2: per-head QKV projection + attention (pipelined)
    for (int hd = 0; hd < NHH; ++hd) {
        const int col0 = hd * DHH;

        wait_async0(); __syncthreads();                       // Wq(hd) ready in wbuf[cur]
        async_copy_16B(WkT + col0 * 256, wbuf[cur ^ 1], 32768, tid);

        { // Q = X @ Wq_slice + bq, scaled by DH^-0.5
            v8f acc[4];
            init_bias(acc, bq, col0, nl);
            gemm_qkv(Xl, wbuf[cur], acc, wv, nl, lane);
#pragma unroll
            for (int nt = 0; nt < 4; ++nt)
#pragma unroll
                for (int r = 0; r < 8; ++r)
                    Qh[(wv * 16 + r + hl * 8) * 64 + nt * 16 + nl] = f2bf(acc[nt][r] * 0.125f);
        }
        wait_async0(); __syncthreads();                       // Wk ready; wbuf[cur] free
        async_copy_16B(WvT + col0 * 256, wbuf[cur], 32768, tid);

        { // K = X @ Wk_slice + bk
            v8f acc[4];
            init_bias(acc, bk, col0, nl);
            gemm_qkv(Xl, wbuf[cur ^ 1], acc, wv, nl, lane);
#pragma unroll
            for (int nt = 0; nt < 4; ++nt)
#pragma unroll
                for (int r = 0; r < 8; ++r)
                    Kh[(wv * 16 + r + hl * 8) * 64 + nt * 16 + nl] = f2bf(acc[nt][r]);
        }
        wait_async0(); __syncthreads();                       // Wv ready; Kh visible

        { // V = X @ Wv_slice + bv, stored transposed Vt[d][h]
            v8f acc[4];
            init_bias(acc, bv, col0, nl);
            gemm_qkv(Xl, wbuf[cur], acc, wv, nl, lane);
#pragma unroll
            for (int nt = 0; nt < 4; ++nt)
#pragma unroll
                for (int r = 0; r < 8; ++r)
                    Vt[(nt * 16 + nl) * 128 + wv * 16 + r + hl * 8] = f2bf(acc[nt][r]);
        }
        __syncthreads();                                      // Vt visible; X reads done

        // Prefetch next tile during the attention phase.
        if (hd < NHH - 1) async_copy_16B(WqT + (col0 + DHH) * 256, wbuf[cur ^ 1], 32768, tid);
        else              async_copy_16B(WoT, wobuf[0], 65536, tid);   // X region now free

        // ---- Scores: S = Qh @ Kh^T ; wave wv owns 16x128 row-block wv.
        v8f s[8];
#pragma unroll
        for (int nt = 0; nt < 8; ++nt)
#pragma unroll
            for (int i = 0; i < 8; ++i) s[nt][i] = 0.0f;
        for (int kt = 0; kt < 2; ++kt) {
            v16bf a = ldsA(Qh, 64, wv * 16 + nl, kt * 32, lane);
#pragma unroll
            for (int nt = 0; nt < 8; ++nt) {
                v16bf b = ldsB(Kh, 64, nt * 16 + nl, kt * 32, lane);   // Kh rows = B cols
                s[nt] = wmma_bf16(a, b, s[nt]);
            }
        }
        __syncthreads();   // all waves done reading Qh/Kh before P (alias) is written

        // ---- Softmax over key axis (128 = 8 ntiles x 16 lanes of same hl group)
        bf16_t* Pw = P + wv * 16 * 128;
#pragma unroll
        for (int r = 0; r < 8; ++r) {
            float mx = -3.0e38f;
#pragma unroll
            for (int nt = 0; nt < 8; ++nt) mx = fmaxf(mx, s[nt][r]);
#pragma unroll
            for (int off = 8; off >= 1; off >>= 1)
                mx = fmaxf(mx, __shfl_xor(mx, off, 32));   // stays within hl half
            float sum = 0.0f;
#pragma unroll
            for (int nt = 0; nt < 8; ++nt) {
                float e = __expf(s[nt][r] - mx);
                s[nt][r] = e;
                sum += e;
            }
#pragma unroll
            for (int off = 8; off >= 1; off >>= 1)
                sum += __shfl_xor(sum, off, 32);
            float inv = __frcp_rn(sum);
#pragma unroll
            for (int nt = 0; nt < 8; ++nt)
                Pw[(r + hl * 8) * 128 + nt * 16 + nl] = f2bf(s[nt][r] * inv);
        }

        // ---- O = P @ V : [16x128]@[128x64] per wave; B = Vt column-major
        v8f o[4];
#pragma unroll
        for (int nt = 0; nt < 4; ++nt)
#pragma unroll
            for (int i = 0; i < 8; ++i) o[nt][i] = 0.0f;
        for (int kt = 0; kt < 4; ++kt) {
            v16bf a = ldsA(Pw, 128, nl, kt * 32, lane);     // same-wave P, DS in-order
#pragma unroll
            for (int nt = 0; nt < 4; ++nt) {
                v16bf b = ldsB(Vt, 128, nt * 16 + nl, kt * 32, lane);
                o[nt] = wmma_bf16(a, b, o[nt]);
            }
        }
#pragma unroll
        for (int nt = 0; nt < 4; ++nt)
#pragma unroll
            for (int r = 0; r < 8; ++r)
                ctx[(wv * 16 + r + hl * 8) * 512 + col0 + nt * 16 + nl] = f2bf(o[nt][r]);

        cur ^= 1;   // next head's Wq was prefetched into the other buffer
    }

    // ---- Stage 3: out = ctx @ Wo + bo, double-buffered 64KB Wo tiles.
    for (int nc = 0; nc < 4; ++nc) {
        wait_async0(); __syncthreads();                      // tile nc ready
        if (nc < 3) async_copy_16B(WoT + (size_t)(nc + 1) * 64 * 512,
                                   wobuf[(nc + 1) & 1], 65536, tid);

        const bf16_t* Wot = wobuf[nc & 1];
        v8f acc[4];
        init_bias(acc, bo, nc * 64, nl);
        for (int kt = 0; kt < 16; ++kt) {
            v16bf a = ldsA(ctx, 512, wv * 16 + nl, kt * 32, lane);
#pragma unroll
            for (int nt = 0; nt < 4; ++nt) {
                v16bf b = ldsB(Wot, 512, nt * 16 + nl, kt * 32, lane);
                acc[nt] = wmma_bf16(a, b, acc[nt]);
            }
        }
#pragma unroll
        for (int nt = 0; nt < 4; ++nt)
#pragma unroll
            for (int r = 0; r < 8; ++r) {
                int m = wv * 16 + r + hl * 8;
                int gc = nc * 64 + nt * 16 + nl;
                out[(((size_t)bx * HH + m) * WW + wx) * CC + gc] = acc[nt][r];
            }
    }
}

extern "C" void kernel_launch(void* const* d_in, const int* in_sizes, int n_in,
                              void* d_out, int out_size, void* d_ws, size_t ws_size,
                              hipStream_t stream) {
    const float* x  = (const float*)d_in[0];
    const float* Wq = (const float*)d_in[1];
    const float* bq = (const float*)d_in[2];
    const float* Wk = (const float*)d_in[3];
    const float* bk = (const float*)d_in[4];
    const float* Wv = (const float*)d_in[5];
    const float* bv = (const float*)d_in[6];
    const float* Wo = (const float*)d_in[7];
    const float* bo = (const float*)d_in[8];
    float* out = (float*)d_out;

    // Workspace: bf16 transposed weights, 4 x 131072 elems = 1 MB.
    bf16_t* WqT = (bf16_t*)d_ws;
    bf16_t* WkT = WqT + 131072;
    bf16_t* WvT = WkT + 131072;
    bf16_t* WoT = WvT + 131072;

    convert_weights<<<dim3(2048), dim3(256), 0, stream>>>(Wq, Wk, Wv, Wo, WqT, WkT, WvT, WoT);

    // LDS: X(64K) + ctx(128K) + Wt0(32K) + Wt1(32K) + Qh(16K) + Kh(16K) + Vt(16K) = 304 KB
    size_t shmem = (size_t)(128 * 256 + 128 * 512 + 2 * 64 * 256 + 3 * 128 * 64) * sizeof(bf16_t);
    axial_mha_fused<<<dim3(BB * WW), dim3(256), shmem, stream>>>(
        x, WqT, WkT, WvT, WoT, bq, bk, bv, bo, out);
}
